// SigLipLoss_4758823764366
// MI455X (gfx1250) — matrix operand order repaired and checked by reading, not compiled, not run on hardware
//
#include <hip/hip_runtime.h>
#include <hip/hip_bf16.h>

typedef __attribute__((ext_vector_type(16))) __bf16 bf16x16;
typedef __attribute__((ext_vector_type(8)))  float  v8f;

#define TM     128   // block tile rows (image)
#define TN     256   // block tile cols (text)
#define KSTEP  32
#define DDIM   768
#define NDIM   16384
#define LDSTR  40    // ushorts per LDS row (32 data + 8 pad) -> 80B, 16B-aligned rows

// ---------------------------------------------------------------------------
// fp32 -> bf16 (round-to-nearest-even), 8 elements per thread
// ---------------------------------------------------------------------------
__global__ __launch_bounds__(256) void cvt_bf16_kernel(
    const float* __restrict__ img, const float* __restrict__ txt,
    unsigned short* __restrict__ abf, unsigned short* __restrict__ bbf,
    size_t nd)
{
    size_t base = ((size_t)blockIdx.x * 256 + threadIdx.x) * 8;
    const float* src;
    unsigned short* dst;
    if (base < nd) { src = img + base;        dst = abf + base; }
    else           { src = txt + (base - nd); dst = bbf + (base - nd); }

    float4 f0 = ((const float4*)src)[0];
    float4 f1 = ((const float4*)src)[1];
    float ff[8] = {f0.x, f0.y, f0.z, f0.w, f1.x, f1.y, f1.z, f1.w};
    unsigned int p[4];
#pragma unroll
    for (int i = 0; i < 4; ++i) {
        unsigned int lo = __float_as_uint(ff[2 * i]);
        unsigned int hi = __float_as_uint(ff[2 * i + 1]);
        lo += 0x7FFFu + ((lo >> 16) & 1u);
        hi += 0x7FFFu + ((hi >> 16) & 1u);
        p[i] = (lo >> 16) | (hi & 0xFFFF0000u);
    }
    uint4 o; o.x = p[0]; o.y = p[1]; o.z = p[2]; o.w = p[3];
    *(uint4*)dst = o;
}

// ---------------------------------------------------------------------------
// helpers
// ---------------------------------------------------------------------------
static __device__ inline bf16x16 mk_frag(uint4 lo, uint4 hi) {
    union { unsigned int u[8]; bf16x16 v; } x;
    x.u[0] = lo.x; x.u[1] = lo.y; x.u[2] = lo.z; x.u[3] = lo.w;
    x.u[4] = hi.x; x.u[5] = hi.y; x.u[6] = hi.z; x.u[7] = hi.w;
    return x.v;
}

static __device__ inline float softplus_f(float x) {
    // log(1+exp(x)), numerically stable
    return fmaxf(x, 0.0f) + __logf(1.0f + __expf(-fabsf(x)));
}

// async 16B copy: global memory -> LDS, no VGPR staging (ASYNCcnt-tracked)
static __device__ inline void async_copy_b128(unsigned lds_addr, const void* g) {
    unsigned long long ga = (unsigned long long)(uintptr_t)g;
    asm volatile("global_load_async_to_lds_b128 %0, %1, off"
                 :: "v"(lds_addr), "v"(ga) : "memory");
}

// ---------------------------------------------------------------------------
// fused bf16 WMMA GEMM (I @ T^T) + siglip softplus reduction, 128x256 tile,
// 8 waves in 2x4 layout, each wave computes a 64x64 subtile (4x4 fragments)
// ---------------------------------------------------------------------------
__global__ __launch_bounds__(256) void siglip_gemm_loss_kernel(
    const unsigned short* __restrict__ abf,  // [N, D] bf16 image features
    const unsigned short* __restrict__ bbf,  // [N, D] bf16 text features
    const float* __restrict__ scale_p, const float* __restrict__ bias_p,
    float* __restrict__ partial)
{
    __shared__ unsigned short shA[TM * LDSTR];   // 128 x 32 (+pad)
    __shared__ unsigned short shB[TN * LDSTR];   // 256 x 32 (+pad)
    __shared__ float wsum[8];

    const int t     = threadIdx.x;
    const int lane  = t & 31;
    const int w     = t >> 5;          // wave 0..7
    const int sub   = lane >> 4;       // half-wave 0/1
    const int l16   = lane & 15;
    const int waveM = w & 1;           // 2 wave rows  -> 64 rows each
    const int waveN = w >> 1;          // 4 wave cols  -> 64 cols each
    const int rowBase = blockIdx.y * TM;
    const int colBase = blockIdx.x * TN;

    // generic LDS pointers: low 32 bits == wave-relative LDS byte address
    const unsigned shAbase = (unsigned)(uintptr_t)&shA[0];
    const unsigned shBbase = (unsigned)(uintptr_t)&shB[0];

    v8f c[4][4] = {};                  // 16 accumulator fragments (16x16 f32)

    for (int kt = 0; kt < DDIM; kt += KSTEP) {
        __syncthreads();               // previous iteration's LDS reads done

        // ---- stage 384 rows x 64B via async global->LDS (6 x 16B / thread) --
#pragma unroll
        for (int j = 0; j < 6; ++j) {
            int e   = t + 256 * j;     // 16B-chunk id 0..1535
            int row = e >> 2;          // tile row 0..383
            int q   = (e & 3) * 16;    // byte offset within 64B K-slice row
            if (j < 2) {               // rows 0..127 -> A tile
                const char* g = (const char*)abf +
                    ((size_t)(rowBase + row) * DDIM + kt) * 2 + q;
                async_copy_b128(shAbase + row * (LDSTR * 2) + q, g);
            } else {                   // rows 128..383 -> B tile
                int br = row - TM;
                const char* g = (const char*)bbf +
                    ((size_t)(colBase + br) * DDIM + kt) * 2 + q;
                async_copy_b128(shBbase + br * (LDSTR * 2) + q, g);
            }
        }
        asm volatile("s_wait_asynccnt 0x0" ::: "memory");
        __syncthreads();               // all waves' async LDS writes visible

        // ---- B fragments (K=32 x N=16): half-waves hold K 0..15 / 16..31 ----
        bf16x16 bfr[4];
#pragma unroll
        for (int fn = 0; fn < 4; ++fn) {
            int br = waveN * 64 + fn * 16 + l16;   // text row within tile
            const unsigned short* p = &shB[br * LDSTR + sub * 16];
            uint4 lo = ((const uint4*)p)[0];
            uint4 hi = ((const uint4*)p)[1];
            bfr[fn] = mk_frag(lo, hi);
        }

        // ---- A fragments (M=16 x K=32): halves hold K 0..15 / 16..31
#pragma unroll
        for (int fm = 0; fm < 4; ++fm) {
            int ar = waveM * 64 + fm * 16 + l16;
            uint4 lo = *(const uint4*)&shA[ar * LDSTR + sub * 16];
            uint4 hi = *(const uint4*)&shA[ar * LDSTR + sub * 16 + 8];
            bf16x16 af = mk_frag(lo, hi);
#pragma unroll
            for (int fn = 0; fn < 4; ++fn) {
                c[fm][fn] = __builtin_amdgcn_wmma_f32_16x16x32_bf16(
                    false, af, false, bfr[fn], (short)0, c[fm][fn], false, false);
            }
        }
    }

    // ---- fused siglip epilogue: logit = s*acc + b; softplus(-label*logit) ----
    const float scale = *scale_p;
    const float bias  = *bias_p;
    float s = 0.0f;
#pragma unroll
    for (int fm = 0; fm < 4; ++fm) {
#pragma unroll
        for (int fn = 0; fn < 4; ++fn) {
#pragma unroll
            for (int r = 0; r < 8; ++r) {
                int gi = rowBase + waveM * 64 + fm * 16 + sub * 8 + r;
                int gj = colBase + waveN * 64 + fn * 16 + l16;
                float logit = scale * c[fm][fn][r] + bias;
                float x = (gi == gj) ? -logit : logit;   // -label*logit
                s += softplus_f(x);
            }
        }
    }

    // wave32 reduction, then cross-wave via LDS; deterministic per-block partial
#pragma unroll
    for (int off = 16; off > 0; off >>= 1) s += __shfl_xor(s, off, 32);
    if (lane == 0) wsum[w] = s;
    __syncthreads();
    if (t == 0) {
        float tt = 0.0f;
#pragma unroll
        for (int q = 0; q < 8; ++q) tt += wsum[q];
        partial[blockIdx.y * gridDim.x + blockIdx.x] = tt;
    }
}

// ---------------------------------------------------------------------------
// deterministic final reduction of tile partials -> scalar loss
// ---------------------------------------------------------------------------
__global__ __launch_bounds__(256) void siglip_finalize_kernel(
    const float* __restrict__ partial, float* __restrict__ out, int n)
{
    __shared__ float sm[256];
    float s = 0.0f;
    for (int i = threadIdx.x; i < n; i += 256) s += partial[i];
    sm[threadIdx.x] = s;
    __syncthreads();
    for (int st = 128; st > 0; st >>= 1) {
        if (threadIdx.x < st) sm[threadIdx.x] += sm[threadIdx.x + st];
        __syncthreads();
    }
    if (threadIdx.x == 0) out[0] = sm[0] / (float)NDIM;
}

// ---------------------------------------------------------------------------
extern "C" void kernel_launch(void* const* d_in, const int* in_sizes, int n_in,
                              void* d_out, int out_size, void* d_ws, size_t ws_size,
                              hipStream_t stream) {
    const float* img     = (const float*)d_in[0];
    const float* txt     = (const float*)d_in[1];
    const float* scale_p = (const float*)d_in[2];
    const float* bias_p  = (const float*)d_in[3];
    float* out = (float*)d_out;

    const size_t ND = (size_t)NDIM * DDIM;
    unsigned short* abf = (unsigned short*)d_ws;
    unsigned short* bbf = abf + ND;
    float* partial = (float*)(bbf + ND);
    const int gx = NDIM / TN;                 // 64
    const int gy = NDIM / TM;                 // 128
    const int nTiles = gx * gy;               // 8192

    // 1) fp32 -> bf16 (both matrices), 8 elems/thread
    int cvtBlocks = (int)((2 * ND / 8 + 255) / 256);    // 12288
    cvt_bf16_kernel<<<cvtBlocks, 256, 0, stream>>>(img, txt, abf, bbf, ND);

    // 2) fused WMMA GEMM + siglip reduction per 128x256 tile
    dim3 grid(gx, gy);
    siglip_gemm_loss_kernel<<<grid, 256, 0, stream>>>(abf, bbf, scale_p, bias_p,
                                                      partial);

    // 3) deterministic final reduction -> scalar
    siglip_finalize_kernel<<<1, 256, 0, stream>>>(partial, out, nTiles);
}
